// GNN_27917287424727
// MI455X (gfx1250) — matrix operand order repaired and checked by reading, compiled, not run on hardware
//
#include <hip/hip_runtime.h>
#include <hip/hip_bf16.h>
#include <stddef.h>

// ---------------------------------------------------------------------------
// GCN (2x GCNConv + global_mean_pool) for MI455X / gfx1250.
//
// Math:  B^T @ M  ==  dinv ⊙ (A^T @ (dinv ⊙ M)),  A binary with unit diag.
// All GEMM operands are exactly representable (binary) or tolerant in bf16,
// so every GEMM runs on v_wmma_f32_16x16x32_bf16 with f32 accumulation.
// A-tile staging uses GLOBAL_LOAD_ASYNC_TO_LDS_B128 (ASYNCcnt) when the
// builtin is available; fenced with s_wait_asynccnt before the barrier.
// ---------------------------------------------------------------------------

typedef __attribute__((ext_vector_type(16))) __bf16 v16bf;
typedef __attribute__((ext_vector_type(8)))  float  v8f;
typedef __attribute__((ext_vector_type(4)))  float  vf4;
typedef __attribute__((ext_vector_type(4)))  int    v4i;

static constexpr int NN = 8192;   // nodes (= feature dim)
static constexpr int HH = 4096;   // hidden
static constexpr int NG = 8;      // graphs

#if defined(__AMDGCN__) && __has_builtin(__builtin_amdgcn_global_load_async_to_lds_b128)
#define HAS_ASYNC_LDS 1
#else
#define HAS_ASYNC_LDS 0
#endif

__device__ __forceinline__ void copy16_g2lds(const __bf16* gsrc, __bf16* ldst) {
#if HAS_ASYNC_LDS
  __builtin_amdgcn_global_load_async_to_lds_b128(
      (__attribute__((address_space(1))) v4i*)gsrc,
      (__attribute__((address_space(3))) v4i*)ldst, 0, 0);
#else
  *(vf4*)ldst = *(const vf4*)gsrc;
#endif
}

__device__ __forceinline__ void async_fence() {
#if HAS_ASYNC_LDS
#if __has_builtin(__builtin_amdgcn_s_wait_asynccnt)
  __builtin_amdgcn_s_wait_asynccnt(0);
#else
  asm volatile("s_wait_asynccnt 0" ::: "memory");
#endif
#endif
}

// ---------------------------------------------------------------- utilities
__global__ __launch_bounds__(256) void k_zero_f32(float* p, int n) {
  for (int i = blockIdx.x * 256 + threadIdx.x; i < n; i += gridDim.x * 256)
    p[i] = 0.0f;
}

__global__ __launch_bounds__(256) void k_cvt_bf16(const float* __restrict__ in,
                                                  __bf16* __restrict__ out,
                                                  long n) {
  long stride = (long)gridDim.x * 256;
  for (long i = blockIdx.x * 256 + threadIdx.x; i < n; i += stride)
    out[i] = (__bf16)in[i];
}

// deg[c] = sum_r A[r,c], A = (x!=0) with forced unit diagonal.
__global__ __launch_bounds__(256) void k_deg(const float* __restrict__ x,
                                             float* __restrict__ deg, int n) {
  int t = threadIdx.x;
  float acc[32];
#pragma unroll
  for (int j = 0; j < 32; ++j) acc[j] = 0.0f;
  for (int r = blockIdx.x; r < n; r += gridDim.x) {
    const float* row = x + (size_t)r * n;
#pragma unroll
    for (int j = 0; j < 32; ++j) {
      int c = t + j * 256;
      float v = row[c];
      float a = (v != 0.0f) ? 1.0f : 0.0f;
      if (r == c) a = 1.0f;          // forced self loop
      acc[j] += a;
    }
  }
#pragma unroll
  for (int j = 0; j < 32; ++j) atomicAdd(&deg[t + j * 256], acc[j]);
}

__global__ __launch_bounds__(256) void k_rsqrt(const float* __restrict__ deg,
                                               float* __restrict__ dinv, int n) {
  int i = blockIdx.x * 256 + threadIdx.x;
  if (i < n) dinv[i] = rsqrtf(deg[i]);   // deg >= 1 (self loops)
}

// At[c*n + r] = bf16( (x[r,c]!=0 || r==c) ? 1 : 0 )   (LDS-tiled transpose)
__global__ __launch_bounds__(256) void k_buildAt(const float* __restrict__ x,
                                                 __bf16* __restrict__ At, int n) {
  __shared__ __bf16 tile[32][33];
  int c0 = blockIdx.x * 32, r0 = blockIdx.y * 32;
  int tc = threadIdx.x & 31, tr8 = threadIdx.x >> 5;   // 0..31, 0..7
#pragma unroll
  for (int s = 0; s < 4; ++s) {
    int rr = tr8 + s * 8;
    int r = r0 + rr, c = c0 + tc;
    float v = x[(size_t)r * n + c];
    tile[rr][tc] = (__bf16)(((v != 0.0f) || (r == c)) ? 1.0f : 0.0f);
  }
  __syncthreads();
#pragma unroll
  for (int s = 0; s < 4; ++s) {
    int rr = tr8 + s * 8;
    At[(size_t)(c0 + rr) * n + (r0 + tc)] = tile[tc][rr];
  }
}

// ---------------------------------------------------------------- bf16 GEMM
// C[M,Nc] = A[M,K] @ B[K,Nc], row-major bf16, f32 accumulation via WMMA.
// Epilogue: val = acc * rowscale[r] (+ bias[c]) (relu) -> bf16 or f32.
// Block tile 128x128, BK=64 (two wmma K sub-steps per stage), double-buffered
// LDS (64 KB total), 8 waves each computing a 32x64 sub-tile (2x4 accums).
__global__ __launch_bounds__(256) void
gemm_bf16(const __bf16* __restrict__ A, const __bf16* __restrict__ Bm,
          const float* __restrict__ rowscale, const float* __restrict__ bias,
          __bf16* __restrict__ out_bf, float* __restrict__ out_f32,
          int M, int Nc, int K, int do_relu) {
  constexpr int BM = 128, BN = 128, BK = 64, BKP = 64;
  __shared__ __bf16 As[2][BM * BK];    // A tile, row-major, K contiguous
  __shared__ __bf16 Bs[2][BN * BKP];   // B tile TRANSPOSED: Bs[n][k], K contig

  const int tid = threadIdx.x;
  const int lane = tid & 31;
  const int w = tid >> 5;              // 8 waves
  const int wm = w & 3;                // 4 waves along M  (32 rows each)
  const int wn = w >> 2;               // 2 waves along N  (64 cols each)
  const int bm0 = blockIdx.y * BM;
  const int bn0 = blockIdx.x * BN;

  v8f acc[2][4];
#pragma unroll
  for (int mi = 0; mi < 2; ++mi)
#pragma unroll
    for (int ni = 0; ni < 4; ++ni)
#pragma unroll
      for (int e = 0; e < 8; ++e) acc[mi][ni][e] = 0.0f;

  // A tile: 128 rows x 64 elems (128 B); 2 threads per row, 4x16B per thread.
  auto stageA = [&](int buf, int k0) {
    int row = tid >> 1, half = tid & 1;
    const __bf16* src = A + (size_t)(bm0 + row) * K + k0 + half * 32;
    __bf16* dst = &As[buf][row * BK + half * 32];
#pragma unroll
    for (int ch = 0; ch < 4; ++ch)
      copy16_g2lds(src + ch * 8, dst + ch * 8);
  };
  // B tile: 64 K-rows x 128 cols, stored transposed (K contiguous).
  auto stageB = [&](int buf, int k0) {
#pragma unroll
    for (int it = 0; it < 4; ++it) {
      int idx = tid + it * 256;                 // 1024 (k, colgroup) pairs
      int k = idx & 63;
      int n0 = (idx >> 6) * 8;
      const __bf16* src = Bm + (size_t)(k0 + k) * Nc + bn0 + n0;
      union { vf4 f; __bf16 h[8]; } u;
      u.f = *(const vf4*)src;
#pragma unroll
      for (int j = 0; j < 8; ++j) Bs[buf][(n0 + j) * BKP + k] = u.h[j];
      if (k0 + 2 * BK < K)
        __builtin_prefetch(src + (size_t)2 * BK * Nc, 0, 1);  // global_prefetch_b8
    }
  };

  stageA(0, 0);
  stageB(0, 0);
  async_fence();
  __syncthreads();

  const int nsteps = K / BK;
  for (int s = 0; s < nsteps; ++s) {
    int cur = s & 1;
    if (s + 1 < nsteps) { stageA(cur ^ 1, (s + 1) * BK); stageB(cur ^ 1, (s + 1) * BK); }

    // ISA wave32 fragment layouts (CDNA5 ISA §7.12.2):
    // A 16x32: lane&15 = row M; lanes<16 hold K 0-7 & 16-23, lanes>=16 hold
    //          K 8-15 & 24-31  (two 16B chunks per lane)
    // B 32x16: lane&15 = col N; lanes<16 hold K 0-7 & 16-23, lanes>=16 hold
    //          K 8-15 & 24-31  (two 16B chunks per lane; Bs is stored transposed)
    const int mrow = lane & 15;
    const int koff = (lane >> 4) * 8;
    const int kbase = (lane >> 4) * 16;
#pragma unroll
    for (int ks = 0; ks < 2; ++ks) {            // two 16x16x32 K sub-steps
      union { v16bf v; vf4 f[2]; } afr[2], bfr[4];
#pragma unroll
      for (int mi = 0; mi < 2; ++mi) {
        const __bf16* p = &As[cur][(wm * 32 + mi * 16 + mrow) * BK + ks * 32];
        afr[mi].f[0] = *(const vf4*)(p + koff);
        afr[mi].f[1] = *(const vf4*)(p + koff + 16);
      }
#pragma unroll
      for (int ni = 0; ni < 4; ++ni) {
        const __bf16* p = &Bs[cur][(wn * 64 + ni * 16 + mrow) * BKP + ks * 32];
        bfr[ni].f[0] = *(const vf4*)(p + koff);
        bfr[ni].f[1] = *(const vf4*)(p + koff + 16);
      }
#pragma unroll
      for (int mi = 0; mi < 2; ++mi)
#pragma unroll
        for (int ni = 0; ni < 4; ++ni)
          acc[mi][ni] = __builtin_amdgcn_wmma_f32_16x16x32_bf16(
              false, afr[mi].v, false, bfr[ni].v, (short)0, acc[mi][ni],
              false, false);
    }
    async_fence();
    __syncthreads();
  }

  // Epilogue. C/D layout: VGPR v -> row v + 8*(lane>=16), col = lane&15.
  const int nlow = lane & 15;
  const int mhi = (lane >> 4) * 8;
#pragma unroll
  for (int mi = 0; mi < 2; ++mi) {
#pragma unroll
    for (int ni = 0; ni < 4; ++ni) {
      int c = bn0 + wn * 64 + ni * 16 + nlow;
      float bv = bias ? bias[c] : 0.0f;
#pragma unroll
      for (int v = 0; v < 8; ++v) {
        int r = bm0 + wm * 32 + mi * 16 + mhi + v;
        float val = acc[mi][ni][v] * rowscale[r] + bv;
        if (do_relu) val = fmaxf(val, 0.0f);
        if (out_f32) out_f32[(size_t)r * Nc + c] = val;
        else         out_bf [(size_t)r * Nc + c] = (__bf16)val;
      }
    }
  }
}

// -------------------------------------------------------------- mean pool
// out[g, c] = sum_{r: batch[r]==g} h2[r, c] / max(cnt[g], 1)
__global__ __launch_bounds__(256) void k_pool(const float* __restrict__ h2,
                                              const int* __restrict__ batch,
                                              float* __restrict__ out, int n) {
  __shared__ int sb[NN];
  __shared__ int scnt[NG];
  int t = threadIdx.x;
  if (t < NG) scnt[t] = 0;
  __syncthreads();
  for (int i = t; i < n; i += 256) {
    int g = batch[i];
    sb[i] = g;
    atomicAdd(&scnt[g], 1);
  }
  __syncthreads();

  int c = blockIdx.x * 256 + t;
  float acc[NG];
#pragma unroll
  for (int q = 0; q < NG; ++q) acc[q] = 0.0f;
  for (int r = 0; r < n; ++r) {
    float v = h2[(size_t)r * n + c];
    int g = sb[r];
#pragma unroll
    for (int q = 0; q < NG; ++q) acc[q] += (g == q) ? v : 0.0f;
  }
#pragma unroll
  for (int q = 0; q < NG; ++q)
    out[(size_t)q * n + c] = acc[q] / fmaxf((float)scnt[q], 1.0f);
}

// ---------------------------------------------------------------- launcher
extern "C" void kernel_launch(void* const* d_in, const int* in_sizes, int n_in,
                              void* d_out, int out_size, void* d_ws,
                              size_t ws_size, hipStream_t stream) {
  (void)in_sizes; (void)n_in; (void)out_size; (void)ws_size;
  const float* x  = (const float*)d_in[0];   // [N,N]
  const float* W1 = (const float*)d_in[1];   // [N,H]
  const float* b1 = (const float*)d_in[2];   // [H]
  const float* W2 = (const float*)d_in[3];   // [H,N]
  const float* b2 = (const float*)d_in[4];   // [N]
  const int* batch = (const int*)d_in[5];    // [N]
  float* out = (float*)d_out;                // [8,N]

  const long N = NN, H = HH;

  // Workspace layout (~512 MB + 64 KB), with aggressive reuse:
  char* ws = (char*)d_ws;
  float*  deg  = (float*)(ws);                               // 32 KB
  float*  dinv = (float*)(ws + 32 * 1024);                   // 32 KB
  __bf16* At   = (__bf16*)(ws + 64 * 1024);                  // 128 MB
  __bf16* xbf  = At + N * N;                                 // 128 MB (-> S2)
  __bf16* W1b  = xbf + N * N;                                //  64 MB
  __bf16* W2b  = W1b + N * H;                                //  64 MB
  __bf16* S1   = W2b + H * N;                                //  64 MB
  __bf16* hb   = S1 + N * H;                                 //  64 MB
  float*  h2   = (float*)W1b;   // reuses W1b..hb (256 MB), dead by GEMM4

  // 1) degree / dinv
  k_zero_f32<<<32, 256, 0, stream>>>(deg, NN);
  k_deg<<<64, 256, 0, stream>>>(x, deg, NN);
  k_rsqrt<<<32, 256, 0, stream>>>(deg, dinv, NN);

  // 2) bf16 operands (x and A^T are binary -> exact in bf16)
  k_cvt_bf16<<<4096, 256, 0, stream>>>(x, xbf, N * N);
  k_cvt_bf16<<<2048, 256, 0, stream>>>(W1, W1b, N * H);
  k_cvt_bf16<<<2048, 256, 0, stream>>>(W2, W2b, H * N);
  k_buildAt<<<dim3(NN / 32, NN / 32), 256, 0, stream>>>(x, At, NN);

  // 3) GEMM chain (grid = (cols/128, rows/128))
  // S1 = dinv ⊙ (x @ W1)
  gemm_bf16<<<dim3(HH / 128, NN / 128), 256, 0, stream>>>(
      xbf, W1b, dinv, nullptr, S1, nullptr, NN, HH, NN, 0);
  // h = relu(dinv ⊙ (A^T @ S1) + b1)
  gemm_bf16<<<dim3(HH / 128, NN / 128), 256, 0, stream>>>(
      At, S1, dinv, b1, hb, nullptr, NN, HH, NN, 1);
  // S2 = dinv ⊙ (h @ W2)   (overwrites xbf)
  gemm_bf16<<<dim3(NN / 128, NN / 128), 256, 0, stream>>>(
      hb, W2b, dinv, nullptr, xbf, nullptr, NN, NN, HH, 0);
  // h2 = dinv ⊙ (A^T @ S2) + b2   (f32, overwrites W1b..hb region)
  gemm_bf16<<<dim3(NN / 128, NN / 128), 256, 0, stream>>>(
      At, xbf, dinv, b2, nullptr, h2, NN, NN, NN, 0);

  // 4) segment mean pool
  k_pool<<<NN / 256, 256, 0, stream>>>(h2, batch, out, NN);
}
